// AdaptiveMoELayer_74577812127931
// MI455X (gfx1250) — matrix operand order repaired and checked by reading, not compiled, not run on hardware
//
#include <hip/hip_runtime.h>

#define DEV __device__ __forceinline__

typedef __attribute__((ext_vector_type(16))) _Float16 v16h;
typedef __attribute__((ext_vector_type(8)))  _Float16 v8h;
typedef __attribute__((ext_vector_type(8)))  float    v8f;

namespace {
constexpr int kB = 4, kS = 2048, kD = 1024, kF = 4096, kE = 4;
constexpr int kM  = kB * kS;      // 8192 tokens
constexpr int kKK = kE * kF;      // 16384 (stacked K for GEMM2)
constexpr int kMc = 2048;         // token chunk (H buffer = kMc*kKK f16 = 64 MB)

constexpr int kLdsRow = 40;               // halves per A row (32 data + 8 pad)
constexpr int kLdsBuf = 128 * kLdsRow;    // 5120 halves per K-step buffer
constexpr unsigned kLdsBufB = kLdsBuf * 2; // bytes per buffer (10240)
}

union HV16 { v16h v; v8h h[2]; };

DEV v16h load_frag2(const _Float16* __restrict__ lanePtr) {
  HV16 r;
  r.h[0] = *(const v8h*)(lanePtr);
  r.h[1] = *(const v8h*)(lanePtr + 16);
  return r.v;
}

DEV v8f wmma16(v16h a, v16h b, v8f c) {
  return __builtin_amdgcn_wmma_f32_16x16x32_f16(false, a, false, b,
                                                (short)0, c, false, false);
}

// Async DMA: each lane copies 16B global -> LDS; tracked with ASYNCcnt.
DEV void async_to_lds_b128(unsigned ldsAddr, const _Float16* g) {
  asm volatile("global_load_async_to_lds_b128 %0, %1, off"
               :: "v"(ldsAddr), "v"((unsigned long long)g)
               : "memory");
}
DEV void wait_asynccnt_le4() {
  asm volatile("s_wait_asynccnt 4" ::: "memory");
}

// ---------------------------------------------------------------------------
// 1) Per-token uncertainty head, expert coefficients, x -> f16
// ---------------------------------------------------------------------------
__global__ __launch_bounds__(256) void moe_prep(
    const float* __restrict__ x, const float* __restrict__ Wu,
    const float* __restrict__ bu, float* __restrict__ coef,
    _Float16* __restrict__ Xh) {
  const int t    = (blockIdx.x * blockDim.x + threadIdx.x) >> 5;  // token
  const int lane = threadIdx.x & 31;
  if (t >= kM) return;
  const float* row = x + (size_t)t * kD;
  float s = 0.f;
  #pragma unroll 4
  for (int i = lane; i < kD; i += 32) {
    const float xv = row[i];
    Xh[(size_t)t * kD + i] = (_Float16)xv;
    s += xv * Wu[i];
  }
  #pragma unroll
  for (int off = 16; off; off >>= 1) s += __shfl_xor(s, off, 32);
  const float u = 1.f / (1.f + __expf(-(s + bu[0])));
  int k = (int)ceilf(u * (float)kE);
  k = k < 1 ? 1 : (k > kE ? kE : k);
  const int sidx = t & (kS - 1);                 // sequence position
  if (lane < kE) {
    const int i = lane + 1;                      // pass index 1..E
    const int e = (sidx + i - 1) & (kE - 1);     // expert chosen at pass i
    const float w = (k >= i) ? (u / (float)i) : 0.f;
    coef[(size_t)t * kE + e] = w;                // each e written exactly once
  }
}

// ---------------------------------------------------------------------------
// 2) One-time repack of a row-major KxN f32 weight matrix into f16 WMMA
//    B-fragment tiles. Tile (kt,nt) at ((nt*Ktiles + kt)*512) halves.
// ---------------------------------------------------------------------------
__global__ __launch_bounds__(256) void moe_swizzle_b(
    const float* __restrict__ src, _Float16* __restrict__ dst,
    int K, int N, long srcBatchStride, long dstBatchStride) {
  const long tid  = (long)blockIdx.x * blockDim.x + threadIdx.x;
  const int  lane = (int)(tid & 31);
  const long tile = tid >> 5;
  const int Ktiles = K >> 5, Ntiles = N >> 4;
  if (tile >= (long)Ktiles * Ntiles) return;
  const int nt = (int)(tile / Ktiles);
  const int kt = (int)(tile % Ktiles);
  const float*  s = src + (long)blockIdx.y * srcBatchStride;
  _Float16*     d = dst + (long)blockIdx.y * dstBatchStride + tile * 512 + lane * 16;
  const int n     = nt * 16 + (lane & 15);
  const int kbase = kt * 32 + ((lane >> 4) ? 8 : 0);
  #pragma unroll
  for (int j = 0; j < 16; ++j) {
    const int k = kbase + (j < 8 ? j : j + 8);
    d[j] = (_Float16)s[(long)k * N + n];
  }
}

// ---------------------------------------------------------------------------
// 3) GEMM1: H[m, e*F+f] = coef[m,e] * relu(Xh[m,:] . W1[e][:,f] + b1[e][f])
//    Block 128x128 (8 waves, 2x4), wave tile 64x32. A tile staged in LDS via
//    async DMA (4-deep ring), B direct from global (pre-swizzled fragments).
// ---------------------------------------------------------------------------
__global__ __launch_bounds__(256) void moe_gemm1(
    const _Float16* __restrict__ Xh, const _Float16* __restrict__ W1p,
    const float* __restrict__ b1, const float* __restrict__ coef,
    _Float16* __restrict__ H, int m0) {
  constexpr int Ktiles = kD / 32;                // 32
  __shared__ _Float16 As[4 * kLdsBuf];           // 40 KB, 4-buffer ring

  const int e    = blockIdx.z;
  const int tid  = threadIdx.x;
  const int wave = tid >> 5, lane = tid & 31;
  const int wm = wave >> 2, wn = wave & 3;       // 2 x 4 wave grid
  const int mBlk = blockIdx.x * 128;
  const int nBlk = blockIdx.y * 128;
  const int mW = mBlk + wm * 64, nW = nBlk + wn * 32;

  // async A copy: thread owns 16B segments tid and tid+256 of the 128x32 tile
  const int segRow = tid >> 2, segPart = tid & 3;
  const _Float16* gA0 = Xh + (size_t)(m0 + mBlk + segRow) * kD + segPart * 8;
  const _Float16* gA1 = gA0 + (size_t)64 * kD;
  const unsigned  lA0 = (unsigned)(unsigned long long)&As[segRow * kLdsRow + segPart * 8];
  const unsigned  lA1 = lA0 + 64 * kLdsRow * 2;

  // B tile pointers (direct global, fragment-linear)
  const _Float16* Bp  = W1p + (size_t)e * ((size_t)Ktiles * (kF / 16) * 512);
  const _Float16* pb0 = Bp + ((size_t)(nW >> 4)) * Ktiles * 512 + lane * 16;
  const _Float16* pb1 = pb0 + (size_t)Ktiles * 512;

  // lane-resolved A fragment read base in LDS
  const _Float16* aRd = &As[(wm * 64 + (lane & 15)) * kLdsRow + ((lane >> 4) << 3)];

  // prologue: tiles 0 and 1 in flight
  async_to_lds_b128(lA0 + 0 * kLdsBufB, gA0);
  async_to_lds_b128(lA1 + 0 * kLdsBufB, gA1);
  gA0 += 32; gA1 += 32;
  async_to_lds_b128(lA0 + 1 * kLdsBufB, gA0);
  async_to_lds_b128(lA1 + 1 * kLdsBufB, gA1);
  gA0 += 32; gA1 += 32;

  v8f acc[4][2] = {};
  #pragma unroll 4
  for (int kt = 0; kt < Ktiles; ++kt) {
    const unsigned wb = ((kt + 2) & 3) * kLdsBufB;
    async_to_lds_b128(lA0 + wb, gA0);
    async_to_lds_b128(lA1 + wb, gA1);
    gA0 += 32; gA1 += 32;
    wait_asynccnt_le4();                 // tile kt landed (in-order retire)
    __syncthreads();                     // ... for every wave in the block
    __builtin_prefetch(pb0 + (size_t)(kt + 4) * 512, 0, 1);
    __builtin_prefetch(pb1 + (size_t)(kt + 4) * 512, 0, 1);
    const v16h b0  = *(const v16h*)(pb0 + (size_t)kt * 512);
    const v16h b1f = *(const v16h*)(pb1 + (size_t)kt * 512);
    const _Float16* ar = aRd + (kt & 3) * kLdsBuf;
    // preload all 4 A fragments (distinct regs) so the 8 WMMAs issue
    // back-to-back behind a single dscnt wait
    v16h a[4];
    #pragma unroll
    for (int im = 0; im < 4; ++im) a[im] = load_frag2(ar + im * 16 * kLdsRow);
    #pragma unroll
    for (int im = 0; im < 4; ++im) {
      acc[im][0] = wmma16(a[im], b0, acc[im][0]);
      acc[im][1] = wmma16(a[im], b1f, acc[im][1]);
    }
  }

  // Epilogue: single base pointer per stream + constant offsets.
  const int ncol = lane & 15;
  const int mhi  = (lane >> 4) * 8;              // C/D: M = v + 8*(lane/16)
  const float* cb = coef + ((size_t)(m0 + mW + mhi)) * kE + e;
  _Float16*    Hb = H + ((size_t)(mW + mhi)) * kKK + (size_t)e * kF + nW + ncol;
  const float bias0 = b1[(size_t)e * kF + nW + ncol];
  const float bias1 = b1[(size_t)e * kF + nW + 16 + ncol];
  #pragma unroll
  for (int im = 0; im < 4; ++im) {
    #pragma unroll
    for (int v = 0; v < 8; ++v) {
      const float c = cb[(im * 16 + v) * kE];
      float h0 = acc[im][0][v] + bias0;
      float h1 = acc[im][1][v] + bias1;
      h0 = h0 > 0.f ? h0 : 0.f;
      h1 = h1 > 0.f ? h1 : 0.f;
      Hb[(size_t)(im * 16 + v) * kKK]      = (_Float16)(h0 * c);
      Hb[(size_t)(im * 16 + v) * kKK + 16] = (_Float16)(h1 * c);
    }
  }
}

// ---------------------------------------------------------------------------
// 4) GEMM2: out[m, d] = H[m,:] . W2_stacked[:, d] + sum_e coef[m,e]*b2[e][d]
//    Single dense GEMM, stacked K = E*F = 16384. Same LDS-A pipeline.
// ---------------------------------------------------------------------------
__global__ __launch_bounds__(256) void moe_gemm2(
    const _Float16* __restrict__ H, const _Float16* __restrict__ W2p,
    const float* __restrict__ b2, const float* __restrict__ coef,
    float* __restrict__ out, int m0) {
  constexpr int Ktiles = kKK / 32;               // 512
  __shared__ _Float16 As[4 * kLdsBuf];           // 40 KB

  const int tid  = threadIdx.x;
  const int wave = tid >> 5, lane = tid & 31;
  const int wm = wave >> 2, wn = wave & 3;
  const int mBlk = blockIdx.x * 128;
  const int nBlk = blockIdx.y * 128;
  const int mW = mBlk + wm * 64, nW = nBlk + wn * 32;

  const int segRow = tid >> 2, segPart = tid & 3;
  const _Float16* gA0 = H + (size_t)(mBlk + segRow) * kKK + segPart * 8;
  const _Float16* gA1 = gA0 + (size_t)64 * kKK;
  const unsigned  lA0 = (unsigned)(unsigned long long)&As[segRow * kLdsRow + segPart * 8];
  const unsigned  lA1 = lA0 + 64 * kLdsRow * 2;

  const _Float16* pb0 = W2p + ((size_t)(nW >> 4)) * Ktiles * 512 + lane * 16;
  const _Float16* pb1 = pb0 + (size_t)Ktiles * 512;

  const _Float16* aRd = &As[(wm * 64 + (lane & 15)) * kLdsRow + ((lane >> 4) << 3)];

  async_to_lds_b128(lA0 + 0 * kLdsBufB, gA0);
  async_to_lds_b128(lA1 + 0 * kLdsBufB, gA1);
  gA0 += 32; gA1 += 32;
  async_to_lds_b128(lA0 + 1 * kLdsBufB, gA0);
  async_to_lds_b128(lA1 + 1 * kLdsBufB, gA1);
  gA0 += 32; gA1 += 32;

  v8f acc[4][2] = {};
  #pragma unroll 4
  for (int kt = 0; kt < Ktiles; ++kt) {
    const unsigned wb = ((kt + 2) & 3) * kLdsBufB;
    async_to_lds_b128(lA0 + wb, gA0);
    async_to_lds_b128(lA1 + wb, gA1);
    gA0 += 32; gA1 += 32;
    wait_asynccnt_le4();
    __syncthreads();
    __builtin_prefetch(pb0 + (size_t)(kt + 4) * 512, 0, 1);
    __builtin_prefetch(pb1 + (size_t)(kt + 4) * 512, 0, 1);
    const v16h b0  = *(const v16h*)(pb0 + (size_t)kt * 512);
    const v16h b1f = *(const v16h*)(pb1 + (size_t)kt * 512);
    const _Float16* ar = aRd + (kt & 3) * kLdsBuf;
    v16h a[4];
    #pragma unroll
    for (int im = 0; im < 4; ++im) a[im] = load_frag2(ar + im * 16 * kLdsRow);
    #pragma unroll
    for (int im = 0; im < 4; ++im) {
      acc[im][0] = wmma16(a[im], b0, acc[im][0]);
      acc[im][1] = wmma16(a[im], b1f, acc[im][1]);
    }
  }

  const int ncol = lane & 15;
  const int mhi  = (lane >> 4) * 8;
  const float* cb = coef + ((size_t)(m0 + mW + mhi)) * kE;
  float*       ob = out + ((size_t)(m0 + mW + mhi)) * kD + nW + ncol;
  const float* b2b = b2 + nW + ncol;
  float wb0[kE], wb1[kE];
  #pragma unroll
  for (int e = 0; e < kE; ++e) {
    wb0[e] = b2b[e * kD];
    wb1[e] = b2b[e * kD + 16];
  }
  #pragma unroll
  for (int im = 0; im < 4; ++im) {
    #pragma unroll
    for (int v = 0; v < 8; ++v) {
      float bias0 = 0.f, bias1 = 0.f;
      #pragma unroll
      for (int e = 0; e < kE; ++e) {
        const float c = cb[(im * 16 + v) * kE + e];
        bias0 += c * wb0[e];
        bias1 += c * wb1[e];
      }
      ob[(size_t)(im * 16 + v) * kD]      = acc[im][0][v] + bias0;
      ob[(size_t)(im * 16 + v) * kD + 16] = acc[im][1][v] + bias1;
    }
  }
}

// ---------------------------------------------------------------------------
extern "C" void kernel_launch(void* const* d_in, const int* in_sizes, int n_in,
                              void* d_out, int out_size, void* d_ws, size_t ws_size,
                              hipStream_t stream) {
  (void)in_sizes; (void)n_in; (void)out_size; (void)ws_size;
  const float* x  = (const float*)d_in[0];
  const float* W1 = (const float*)d_in[1];
  const float* b1 = (const float*)d_in[2];
  const float* W2 = (const float*)d_in[3];
  const float* b2 = (const float*)d_in[4];
  const float* Wu = (const float*)d_in[5];
  const float* bu = (const float*)d_in[6];
  float* out = (float*)d_out;

  // Workspace layout (~145 MB). coef deliberately last: gives slack for the
  // <=128B pipeline over-read past the end of H.
  char* ws = (char*)d_ws;
  _Float16* Xh   = (_Float16*)ws;  ws += (size_t)kM * kD * sizeof(_Float16);      // 16 MB
  _Float16* W1p  = (_Float16*)ws;  ws += (size_t)kE * kD * kF * sizeof(_Float16); // 32 MB
  _Float16* W2p  = (_Float16*)ws;  ws += (size_t)kKK * kD * sizeof(_Float16);     // 32 MB
  _Float16* H    = (_Float16*)ws;  ws += (size_t)kMc * kKK * sizeof(_Float16);    // 64 MB
  float*    coef = (float*)ws;     // 128 KB

  // 1) uncertainty head + coefficients + x -> f16
  moe_prep<<<dim3((kM * 32) / 256), dim3(256), 0, stream>>>(x, Wu, bu, coef, Xh);

  // 2) repack weights into WMMA B-fragment layout (f32 -> f16)
  {
    const long tilesW1 = (long)(kD / 32) * (kF / 16);    // 8192 per expert
    moe_swizzle_b<<<dim3((unsigned)((tilesW1 * 32) / 256), kE), dim3(256), 0, stream>>>(
        W1, W1p, kD, kF, (long)kD * kF, tilesW1 * 512);
    const long tilesW2 = (long)(kKK / 32) * (kD / 16);   // 32768
    moe_swizzle_b<<<dim3((unsigned)((tilesW2 * 32) / 256), 1), dim3(256), 0, stream>>>(
        W2, W2p, kKK, kD, 0, 0);
  }

  // 3) chunked fused MoE MLP: GEMM1 (per expert) then stacked-K GEMM2
  for (int m0 = 0; m0 < kM; m0 += kMc) {
    moe_gemm1<<<dim3(kMc / 128, kF / 128, kE), dim3(256), 0, stream>>>(
        Xh, W1p, b1, coef, H, m0);
    moe_gemm2<<<dim3(kMc / 128, kD / 128), dim3(256), 0, stream>>>(
        H, W2p, b2, coef, out, m0);
  }
}
